// BaGuaModel_88381837017349
// MI455X (gfx1250) — compile-verified, matrix operand order checked by reading, not run on hardware
//
#include <hip/hip_runtime.h>
#include <math.h>

typedef _Float16 v16h_t __attribute__((ext_vector_type(16)));
typedef float    v8f_t  __attribute__((ext_vector_type(8)));
typedef float    f4_t   __attribute__((ext_vector_type(4)));

#define BB 8
#define SS 2048
#define DD 512
#define LL 4
#define KK 64
#define PP 32
#define MEMN 16
#define FFN 2048
#define MROWS (BB*SS)          // 16384
#define PI_F 3.14159265358979323846f

// act: 0 none, 1 gelu(exact/erf), 2 sigmoid, 3 tanh
__device__ __forceinline__ float bg_act(float x, int act)
{
    if (act == 1)      return 0.5f * x * (1.f + erff(x * 0.70710678118654752f));
    else if (act == 2) return 1.f / (1.f + expf(-x));
    else if (act == 3) return tanhf(x);
    return x;
}

// Load one 16x32 f16 operand fragment (A row-major or W^T row-major), ISA layout:
// lane = hi*16 + r; element e -> K = (e>>3)*16 + hi*8 + (e&7)
__device__ __forceinline__ v16h_t bg_load_frag(const float* __restrict__ rowbase,
                                               int k, int hi)
{
    const f4_t* p = (const f4_t*)(rowbase + k + hi * 8);
    const f4_t* q = (const f4_t*)(rowbase + k + 16 + hi * 8);
    f4_t q0 = p[0], q1 = p[1], q2 = q[0], q3 = q[1];
    v16h_t v;
#pragma unroll
    for (int i = 0; i < 4; ++i) {
        v[i]      = (_Float16)q0[i];
        v[4 + i]  = (_Float16)q1[i];
        v[8 + i]  = (_Float16)q2[i];
        v[12 + i] = (_Float16)q3[i];
    }
    return v;
}

__device__ __forceinline__ void bg_store_tile(v8f_t acc, int row0, int col, int hi,
                                              const float* __restrict__ bias,
                                              const float* __restrict__ addsrc,
                                              float* __restrict__ C, int N, int act)
{
    const float bv = bias ? bias[col] : 0.f;
#pragma unroll
    for (int v = 0; v < 8; ++v) {
        int row = row0 + v + 8 * hi;
        float x = acc[v] + bv;
        if (addsrc) x += addsrc[(size_t)row * N + col];
        C[(size_t)row * N + col] = bg_act(x, act);
    }
}

// ---------------------------------------------------------------------------
// Blocked GEMM: C[M,N] = act(A[M,K] @ W[N,K]^T + bias + add).
// 256 threads = 8 waves (4 along M x 2 along N); wave computes 32x32 via
// 4 independent WMMA accumulators (2 A-frags x 2 B-frags per K-step).
// Requires M % 128 == 0, N % 64 == 0, K % 32 == 0.
// ---------------------------------------------------------------------------
__global__ void __launch_bounds__(256)
bg_gemm_f16_2x2(const float* __restrict__ A, int lda,
                const float* __restrict__ W, int ldw,
                const float* __restrict__ bias,
                const float* __restrict__ addsrc,
                float* __restrict__ C,
                int M, int N, int K, int act)
{
    const int lane = threadIdx.x & 31;
    const int wave = threadIdx.x >> 5;
    const int r    = lane & 15;
    const int hi   = lane >> 4;
    const int wm   = wave & 3;
    const int wn   = wave >> 2;
    const int m0   = blockIdx.y * 128 + wm * 32;
    const int n0   = blockIdx.x * 64 + wn * 32;

    const float* Arow0 = A + (size_t)(m0 + r) * lda;
    const float* Arow1 = A + (size_t)(m0 + 16 + r) * lda;
    const float* Wrow0 = W + (size_t)(n0 + r) * ldw;
    const float* Wrow1 = W + (size_t)(n0 + 16 + r) * ldw;

    v8f_t acc00 = {0.f,0.f,0.f,0.f,0.f,0.f,0.f,0.f};
    v8f_t acc01 = acc00, acc10 = acc00, acc11 = acc00;

    for (int k = 0; k < K; k += 32) {
        v16h_t a0 = bg_load_frag(Arow0, k, hi);
        v16h_t a1 = bg_load_frag(Arow1, k, hi);
        v16h_t b0 = bg_load_frag(Wrow0, k, hi);
        v16h_t b1 = bg_load_frag(Wrow1, k, hi);
        acc00 = __builtin_amdgcn_wmma_f32_16x16x32_f16(false, a0, false, b0,
                                                       (short)0, acc00, false, false);
        acc01 = __builtin_amdgcn_wmma_f32_16x16x32_f16(false, a0, false, b1,
                                                       (short)0, acc01, false, false);
        acc10 = __builtin_amdgcn_wmma_f32_16x16x32_f16(false, a1, false, b0,
                                                       (short)0, acc10, false, false);
        acc11 = __builtin_amdgcn_wmma_f32_16x16x32_f16(false, a1, false, b1,
                                                       (short)0, acc11, false, false);
    }

    bg_store_tile(acc00, m0,      n0 + r,      hi, bias, addsrc, C, N, act);
    bg_store_tile(acc01, m0,      n0 + 16 + r, hi, bias, addsrc, C, N, act);
    bg_store_tile(acc10, m0 + 16, n0 + r,      hi, bias, addsrc, C, N, act);
    bg_store_tile(acc11, m0 + 16, n0 + 16 + r, hi, bias, addsrc, C, N, act);
}

// ---------------------------------------------------------------------------
// Single-tile GEMM for narrow outputs (N = 16): one 16x16 tile per wave.
// ---------------------------------------------------------------------------
__global__ void __launch_bounds__(256)
bg_gemm_f16(const float* __restrict__ A, int lda,
            const float* __restrict__ W, int ldw,
            const float* __restrict__ bias,
            const float* __restrict__ addsrc,
            float* __restrict__ C,
            int M, int N, int K, int act)
{
    const int lane = threadIdx.x & 31;
    const int wave = threadIdx.x >> 5;
    const int r    = lane & 15;
    const int hi   = lane >> 4;
    const int m0   = blockIdx.y * 128 + wave * 16;
    const int n0   = blockIdx.x * 16;
    if (m0 >= M) return;

    const float* Arow = A + (size_t)(m0 + r) * lda;
    const float* Wrow = W + (size_t)(n0 + r) * ldw;

    v8f_t acc = {0.f,0.f,0.f,0.f,0.f,0.f,0.f,0.f};
    for (int k = 0; k < K; k += 32) {
        v16h_t a = bg_load_frag(Arow, k, hi);
        v16h_t b = bg_load_frag(Wrow, k, hi);
        acc = __builtin_amdgcn_wmma_f32_16x16x32_f16(false, a, false, b,
                                                     (short)0, acc, false, false);
    }
    bg_store_tile(acc, m0, n0 + r, hi, bias, addsrc, C, N, act);
}

// ---------------------------------------------------------------------------
// Small / elementwise kernels
// ---------------------------------------------------------------------------
__global__ void bg_embed(const int* __restrict__ ids, const float* __restrict__ emb,
                         const float* __restrict__ pos, float* __restrict__ X,
                         float* __restrict__ PREV)
{
    size_t id = (size_t)blockIdx.x * 256 + threadIdx.x;   // B*S*D
    int d = (int)(id & (DD - 1));
    size_t bs = id >> 9;
    int s = (int)(bs & (SS - 1));
    int tok = ids[bs];
    float v = emb[(size_t)tok * DD + d] + pos[(size_t)s * DD + d];
    X[id] = v;
    PREV[id] = v;
}

// fold cos(res_freq * pi) per output-channel into trigram weight rows
__global__ void bg_scalew(const float* __restrict__ triW, const float* __restrict__ freq,
                          float* __restrict__ out)
{
    size_t id = (size_t)blockIdx.x * 256 + threadIdx.x;   // 512*512
    int c = (int)(id >> 9);
    out[id] = triW[id] * cosf(freq[c] * PI_F);
}

// summary[b,c] = mean_s heads[b,s,c]
__global__ void bg_summary(const float* __restrict__ H, float* __restrict__ summ)
{
    int t = blockIdx.x * 256 + threadIdx.x;               // B*512
    int b = t >> 9, c = t & (DD - 1);
    const float* p = H + (size_t)b * SS * DD + c;
    float s = 0.f;
    for (int ss = 0; ss < SS; ++ss) s += p[(size_t)ss * DD];
    summ[t] = s * (1.f / (float)SS);
}

// pol[b,h,p] = tanh(summary[b,h,:] . pol_W[h,p,:] + pol_b[h,p]), L2-normalized over p
__global__ void bg_pol(const float* __restrict__ summ, const float* __restrict__ polW,
                       const float* __restrict__ polb, float* __restrict__ pol)
{
    int b = blockIdx.x >> 3, h = blockIdx.x & 7;          // grid 64, block 32
    int p = threadIdx.x;
    const float* s = summ + b * DD + h * KK;
    const float* w = polW + (size_t)(h * PP + p) * KK;
    float acc = polb[h * PP + p];
#pragma unroll 8
    for (int k = 0; k < KK; ++k) acc += s[k] * w[k];
    float tv = tanhf(acc);
    float sq = tv * tv;
#pragma unroll
    for (int off = 16; off; off >>= 1) sq += __shfl_xor(sq, off, 32);
    float nrm = fmaxf(sqrtf(sq), 1e-12f);
    pol[(b * 8 + h) * PP + p] = tv / nrm;
}

// coef[b,i,j] = (i==j) ? 0 : 0.1/(1+softplus(MLP(pol_i . pol_j)))
__global__ void bg_coef(const float* __restrict__ pol, const float* __restrict__ w1,
                        const float* __restrict__ b1, const float* __restrict__ w2,
                        const float* __restrict__ b2, float* __restrict__ coef)
{
    int b = blockIdx.x;                                   // grid 8, block 64
    int t = threadIdx.x;
    int i = t >> 3, j = t & 7;
    const float* pi = pol + (size_t)(b * 8 + i) * PP;
    const float* pj = pol + (size_t)(b * 8 + j) * PP;
    float dot = 0.f;
#pragma unroll 8
    for (int p = 0; p < PP; ++p) dot += pi[p] * pj[p];
    float acc = b2[0];
#pragma unroll
    for (int q = 0; q < 16; ++q) {
        float hm = dot * w1[q] + b1[q];
        hm = 0.5f * hm * (1.f + erff(hm * 0.70710678118654752f)); // exact gelu
        acc += hm * w2[q];
    }
    float sp = log1pf(expf(-fabsf(acc))) + fmaxf(acc, 0.f);      // stable softplus
    coef[b * 64 + t] = (i == j) ? 0.f : 0.1f / (1.f + sp);
}

// in-place head mixing: H[b,s,i,k] += sum_j coef[b,i,j]*H[b,s,j,k]
__global__ void bg_mix(float* __restrict__ H, const float* __restrict__ coef)
{
    size_t id = (size_t)blockIdx.x * 256 + threadIdx.x;   // B*S*K
    int k = (int)(id & (KK - 1));
    size_t bs = id >> 6;                                  // b*S + s
    int b = (int)(bs >> 11);
    float* base = H + bs * DD + k;
    float h[8];
#pragma unroll
    for (int j = 0; j < 8; ++j) h[j] = base[(size_t)j * KK];
    const float* cf = coef + b * 64;
    float o[8];
#pragma unroll
    for (int i = 0; i < 8; ++i) {
        float a = h[i];
#pragma unroll
        for (int j = 0; j < 8; ++j) a += cf[i * 8 + j] * h[j];
        o[i] = a;
    }
#pragma unroll
    for (int i = 0; i < 8; ++i) base[(size_t)i * KK] = o[i];
}

// LayerNorm over D=512; one wave per row (16 elems/lane), shfl reductions
__global__ void bg_ln(const float* __restrict__ in, const float* __restrict__ g,
                      const float* __restrict__ beta, float* __restrict__ out, int rows)
{
    int wave = threadIdx.x >> 5, lane = threadIdx.x & 31;
    int row = blockIdx.x * 8 + wave;
    if (row >= rows) return;
    const float* p = in + (size_t)row * DD;
    float v[16];
    float s = 0.f;
#pragma unroll
    for (int i = 0; i < 16; ++i) { v[i] = p[lane + i * 32]; s += v[i]; }
#pragma unroll
    for (int off = 16; off; off >>= 1) s += __shfl_xor(s, off, 32);
    float mean = s * (1.f / (float)DD);
    float s2 = 0.f;
#pragma unroll
    for (int i = 0; i < 16; ++i) { float d = v[i] - mean; s2 += d * d; }
#pragma unroll
    for (int off = 16; off; off >>= 1) s2 += __shfl_xor(s2, off, 32);
    float inv = rsqrtf(s2 * (1.f / (float)DD) + 1e-5f);
    float* q = out + (size_t)row * DD;
#pragma unroll
    for (int i = 0; i < 16; ++i) {
        int c = lane + i * 32;
        q[c] = (v[i] - mean) * inv * g[c] + beta[c];
    }
}

// buffer: x = x + g2 * (g1 * prev); writes new x to X and PREV
__global__ void bg_buffer(const float* __restrict__ xin, const float* __restrict__ g2,
                          const float* __restrict__ g1, float* __restrict__ X,
                          float* __restrict__ PREV)
{
    size_t id = (size_t)blockIdx.x * 256 + threadIdx.x;
    float pv = PREV[id];
    float v = xin[id] + g2[id] * (g1[id] * pv);
    X[id] = v;
    PREV[id] = v;
}

// vbar[d] = mean_r zt_W[r,d]; vbar[512] = mean(zt_b)
__global__ void bg_vbar(const float* __restrict__ ztW, const float* __restrict__ ztb,
                        float* __restrict__ vbar)
{
    int d = threadIdx.x;                                  // block 512
    float s = 0.f;
    for (int r = 0; r < DD; ++r) s += ztW[(size_t)r * DD + d];
    vbar[d] = s * (1.f / (float)DD);
    if (d == 0) {
        float sb = 0.f;
        for (int r = 0; r < DD; ++r) sb += ztb[r];
        vbar[DD] = sb * (1.f / (float)DD);
    }
}

// mval[row] = x[row,:] . vbar + vbar[512]   (one wave per row)
__global__ void bg_mval(const float* __restrict__ X, const float* __restrict__ vbar,
                        float* __restrict__ mval, int rows)
{
    int wave = threadIdx.x >> 5, lane = threadIdx.x & 31;
    int row = blockIdx.x * 8 + wave;
    if (row >= rows) return;
    const float* p = X + (size_t)row * DD;
    float s = 0.f;
#pragma unroll
    for (int i = 0; i < 16; ++i) { int c = lane + i * 32; s += p[c] * vbar[c]; }
#pragma unroll
    for (int off = 16; off; off >>= 1) s += __shfl_xor(s, off, 32);
    if (lane == 0) mval[row] = s + vbar[DD];
}

// sequential memory recurrence (in-place: R -> rw*M); 128 threads = B*MEM lanes
__global__ void bg_scan(float* __restrict__ R, const float* __restrict__ Wg,
                        const float* __restrict__ mval)
{
    int t = threadIdx.x;
    int b = t >> 4, j = t & 15;
    float Ms = 0.f;
    for (int s = 0; s < SS; ++s) {
        size_t row = (size_t)(b * SS + s);
        size_t idx = row * MEMN + j;
        float r = R[idx], w = Wg[idx], mv = mval[row];
        R[idx] = r * Ms;                       // rw * M (M before update)
        Ms = (1.f - w) * Ms + w * mv;
    }
}

// memv[m,d] = RM[m,:] . zp_W[d,:] + zp_b[d]   (K = 16)
__global__ void bg_memv(const float* __restrict__ RM, const float* __restrict__ zpW,
                        const float* __restrict__ zpb, float* __restrict__ out)
{
    size_t id = (size_t)blockIdx.x * 256 + threadIdx.x;   // M*D
    int d = (int)(id & (DD - 1));
    size_t m = id >> 9;
    const float* rm = RM + m * MEMN;
    const float* w = zpW + (size_t)d * MEMN;
    float s = zpb[d];
#pragma unroll
    for (int j = 0; j < MEMN; ++j) s += rm[j] * w[j];
    out[id] = s;
}

// pooled[b,d] = mean_s F[b,s,d]
__global__ void bg_pool(const float* __restrict__ F, float* __restrict__ pooled)
{
    int t = blockIdx.x * 256 + threadIdx.x;               // B*D
    int b = t >> 9, d = t & (DD - 1);
    const float* p = F + (size_t)b * SS * DD + d;
    float s = 0.f;
    for (int ss = 0; ss < SS; ++ss) s += p[(size_t)ss * DD];
    pooled[t] = s * (1.f / (float)SS);
}

__device__ __forceinline__ float bg_block_sum(float v, float* red)
{
    for (int off = 16; off; off >>= 1) v += __shfl_xor(v, off, 32);
    int lane = threadIdx.x & 31, wv = threadIdx.x >> 5;
    if (lane == 0) red[wv] = v;
    __syncthreads();
    float s = 0.f;
    if (threadIdx.x < 32) {
        s = (threadIdx.x < 16) ? red[threadIdx.x] : 0.f;
        for (int off = 8; off; off >>= 1) s += __shfl_xor(s, off, 32);
        if (threadIdx.x == 0) red[0] = s;
    }
    __syncthreads();
    float out = red[0];
    __syncthreads();
    return out;
}

// final: LN(pooled) @ cls_W^T + cls_b -> out[b,4]; one block per batch, 512 threads
__global__ void bg_final(const float* __restrict__ pooled, const float* __restrict__ g,
                         const float* __restrict__ beta, const float* __restrict__ clsW,
                         const float* __restrict__ clsb, float* __restrict__ out)
{
    __shared__ float red[16];
    __shared__ float arr[DD];
    int b = blockIdx.x, t = threadIdx.x;
    float v = pooled[b * DD + t];
    float mean = bg_block_sum(v, red) * (1.f / (float)DD);
    float d = v - mean;
    float var = bg_block_sum(d * d, red) * (1.f / (float)DD);
    arr[t] = d * rsqrtf(var + 1e-5f) * g[t] + beta[t];
    __syncthreads();
    if (t < 4) {
        float acc = clsb[t];
        for (int dd = 0; dd < DD; ++dd) acc += arr[dd] * clsW[t * DD + dd];
        out[b * 4 + t] = acc;
    }
}

// ---------------------------------------------------------------------------
// Host orchestration
// ---------------------------------------------------------------------------
extern "C" void kernel_launch(void* const* d_in, const int* in_sizes, int n_in,
                              void* d_out, int out_size, void* d_ws, size_t ws_size,
                              hipStream_t stream)
{
    (void)in_sizes; (void)n_in; (void)out_size; (void)ws_size;

    const int*   ids     = (const int*)  d_in[0];
    const float* emb     = (const float*)d_in[1];
    const float* pos_emb = (const float*)d_in[2];
    const float* tri_W   = (const float*)d_in[3];
    const float* res_fr  = (const float*)d_in[4];
    const float* pol_W   = (const float*)d_in[5];
    const float* pol_b   = (const float*)d_in[6];
    const float* imp_w1  = (const float*)d_in[7];
    const float* imp_b1  = (const float*)d_in[8];
    const float* imp_w2  = (const float*)d_in[9];
    const float* imp_b2  = (const float*)d_in[10];
    const float* out_W   = (const float*)d_in[11];
    const float* out_b   = (const float*)d_in[12];
    const float* ln1_g   = (const float*)d_in[13];
    const float* ln1_b   = (const float*)d_in[14];
    const float* ffln_g  = (const float*)d_in[15];
    const float* ffln_b  = (const float*)d_in[16];
    const float* ff_W1   = (const float*)d_in[17];
    const float* ff_b1   = (const float*)d_in[18];
    const float* ff_W2   = (const float*)d_in[19];
    const float* ff_b2   = (const float*)d_in[20];
    const float* chg_W   = (const float*)d_in[21];
    const float* chg_b   = (const float*)d_in[22];
    const float* dis_W   = (const float*)d_in[23];
    const float* dis_b   = (const float*)d_in[24];
    const float* zr_W    = (const float*)d_in[25];
    const float* zr_b    = (const float*)d_in[26];
    const float* zw_W    = (const float*)d_in[27];
    const float* zw_b    = (const float*)d_in[28];
    const float* zt_W    = (const float*)d_in[29];
    const float* zt_b    = (const float*)d_in[30];
    const float* zp_W    = (const float*)d_in[31];
    const float* zp_b    = (const float*)d_in[32];
    const float* zo_W    = (const float*)d_in[33];
    const float* zo_b    = (const float*)d_in[34];
    const float* clsln_g = (const float*)d_in[35];
    const float* clsln_b = (const float*)d_in[36];
    const float* cls_W   = (const float*)d_in[37];
    const float* cls_b   = (const float*)d_in[38];
    float* outp = (float*)d_out;

    const size_t MD = (size_t)MROWS * DD;                 // 8,388,608 floats
    float* ws   = (float*)d_ws;
    float* X    = ws;
    float* PREV = ws + MD;
    float* SA   = ws + 2 * MD;
    float* SB   = ws + 3 * MD;
    float* BUF  = ws + 4 * MD;                            // M x FF
    float* SM   = ws + 4 * MD + (size_t)MROWS * FFN;
    float* Wsc    = SM;                  // 512*512
    float* summ   = Wsc + DD * DD;       // B*512
    float* polv   = summ + BB * DD;      // B*8*32
    float* coefv  = polv + BB * 8 * PP;  // B*64
    float* Rb     = coefv + BB * 64;     // M*16
    float* Wgb    = Rb + (size_t)MROWS * MEMN;
    float* mvalb  = Wgb + (size_t)MROWS * MEMN;           // M
    float* vbarb  = mvalb + MROWS;                        // 513
    float* pooled = vbarb + DD + 8;                       // B*512

    const dim3 blk256(256), blk512(512), blk64(64), blk32(32), blk128(128);
    const int ELT_BLOCKS = (int)(MD / 256);               // 32768
    const dim3 g512(DD / 64, MROWS / 128);                // blocked GEMM, N=512
    const dim3 gff(FFN / 64, MROWS / 128);                // blocked GEMM, N=2048
    const dim3 gmem(MEMN / 16, MROWS / 128);              // narrow GEMM, N=16

    // embedding (+pos), also seeds PREV
    bg_embed<<<ELT_BLOCKS, blk256, 0, stream>>>(ids, emb, pos_emb, X, PREV);

    for (int l = 0; l < LL; ++l) {
        const float* triWl = tri_W + (size_t)l * DD * DD;
        const float* frl   = res_fr + (size_t)l * DD;
        const float* polWl = pol_W + (size_t)l * 8 * PP * KK;
        const float* polbl = pol_b + (size_t)l * 8 * PP;
        const float* w1l   = imp_w1 + l * 16;
        const float* b1l   = imp_b1 + l * 16;
        const float* w2l   = imp_w2 + l * 16;
        const float* b2l   = imp_b2 + l;
        const float* outWl = out_W + (size_t)l * DD * DD;
        const float* outbl = out_b + (size_t)l * DD;
        const float* ffW1l = ff_W1 + (size_t)l * FFN * DD;
        const float* ffb1l = ff_b1 + (size_t)l * FFN;
        const float* ffW2l = ff_W2 + (size_t)l * DD * FFN;
        const float* ffb2l = ff_b2 + (size_t)l * DD;

        // trigram weights with resonance folded in, then heads GEMM -> SA
        bg_scalew<<<DD * DD / 256, blk256, 0, stream>>>(triWl, frl, Wsc);
        bg_gemm_f16_2x2<<<g512, blk256, 0, stream>>>(X, DD, Wsc, DD, nullptr, nullptr,
                                                     SA, MROWS, DD, DD, 0);
        // polarity / impedance coefficients
        bg_summary<<<BB * DD / 256, blk256, 0, stream>>>(SA, summ);
        bg_pol<<<BB * 8, blk32, 0, stream>>>(summ, polWl, polbl, polv);
        bg_coef<<<BB, blk64, 0, stream>>>(polv, w1l, b1l, w2l, b2l, coefv);
        // impedance-gated head mixing (in-place on SA)
        bg_mix<<<(int)((size_t)MROWS * KK / 256), blk256, 0, stream>>>(SA, coefv);
        // out projection + residual, then LN -> SB
        bg_gemm_f16_2x2<<<g512, blk256, 0, stream>>>(SA, DD, outWl, DD, outbl, X,
                                                     SB, MROWS, DD, DD, 0);
        bg_ln<<<MROWS / 8, blk256, 0, stream>>>(SB, ln1_g + l * DD, ln1_b + l * DD, SB, MROWS);
        // FF block: h = LN(x) -> SA; H1 = gelu(h@W1^T+b1) -> BUF; x += H1@W2^T+b2 -> SA
        bg_ln<<<MROWS / 8, blk256, 0, stream>>>(SB, ffln_g + l * DD, ffln_b + l * DD, SA, MROWS);
        bg_gemm_f16_2x2<<<gff, blk256, 0, stream>>>(SA, DD, ffW1l, DD, ffb1l, nullptr,
                                                    BUF, MROWS, FFN, DD, 1);
        bg_gemm_f16_2x2<<<g512, blk256, 0, stream>>>(BUF, FFN, ffW2l, FFN, ffb2l, SB,
                                                     SA, MROWS, DD, FFN, 0);
        // charge/discharge buffer
        bg_gemm_f16_2x2<<<g512, blk256, 0, stream>>>(PREV, DD, chg_W, DD, chg_b, nullptr,
                                                     SB, MROWS, DD, DD, 2);   // G1
        bg_gemm_f16_2x2<<<g512, blk256, 0, stream>>>(SA, DD, dis_W, DD, dis_b, nullptr,
                                                     BUF, MROWS, DD, DD, 2);  // G2
        bg_buffer<<<ELT_BLOCKS, blk256, 0, stream>>>(SA, BUF, SB, X, PREV);
    }

    // ---- sequential memory stage (restructured as GEMMs + tiny scan) ----
    bg_gemm_f16<<<gmem, blk256, 0, stream>>>(X, DD, zr_W, DD, zr_b, nullptr,
                                             Rb, MROWS, MEMN, DD, 2);
    bg_gemm_f16<<<gmem, blk256, 0, stream>>>(X, DD, zw_W, DD, zw_b, nullptr,
                                             Wgb, MROWS, MEMN, DD, 2);
    bg_vbar<<<1, blk512, 0, stream>>>(zt_W, zt_b, vbarb);
    bg_mval<<<MROWS / 8, blk256, 0, stream>>>(X, vbarb, mvalb, MROWS);
    bg_scan<<<1, blk128, 0, stream>>>(Rb, Wgb, mvalb);          // Rb becomes rw*M
    bg_memv<<<ELT_BLOCKS, blk256, 0, stream>>>(Rb, zp_W, zp_b, SA);
    // fused = tanh(X @ zo_W[:, :D]^T + memv @ zo_W[:, D:]^T + zo_b)
    bg_gemm_f16_2x2<<<g512, blk256, 0, stream>>>(X, DD, zo_W, 2 * DD, nullptr, nullptr,
                                                 SB, MROWS, DD, DD, 0);
    bg_gemm_f16_2x2<<<g512, blk256, 0, stream>>>(SA, DD, zo_W + DD, 2 * DD, zo_b, SB,
                                                 X, MROWS, DD, DD, 3);
    // pool + classifier head
    bg_pool<<<BB * DD / 256, blk256, 0, stream>>>(X, pooled);
    bg_final<<<BB, blk512, 0, stream>>>(pooled, clsln_g, clsln_b, cls_W, cls_b, outp);
}
